// SS2DWrapper2D_50843822850449
// MI455X (gfx1250) — matrix operand order repaired and checked by reading, not compiled, not run on hardware
//
#include <hip/hip_runtime.h>
#include <hip/hip_bf16.h>
#include <math.h>

// ---------------- problem constants ----------------
#define BB   4
#define HH   48
#define WW   48
#define CC   128
#define DD   256          // d_inner
#define NN   16           // d_state
#define RR   8            // dt_rank
#define KK   4            // directions
#define LL   (HH * WW)    // 2304
#define CPROJ 40          // R + 2N
#define SCHUNK 16         // scan chunks
#define CLEN  (LL / SCHUNK)   // 144

typedef __attribute__((ext_vector_type(16))) __bf16 v16bf;
typedef __attribute__((ext_vector_type(8)))  float  v8f;

// ---------------- WMMA fragment helpers (CDNA5 16x16x32 bf16) ----------------
// A-matrix 16x32 bf16 (ISA 7.12.2): lane holds row M = lane%16.
// VGPR pair p (p=0..7), element e=2p+{0,1}:
//   K = (p>=4 ? 16 : 0) + (p&3)*2 + (lane>=16 ? 8 : 0) + (e&1)
__device__ __forceinline__ v16bf load_a_frag(const float* __restrict__ row,
                                             int kk, int half) {
  v16bf a;
#pragma unroll
  for (int e = 0; e < 16; ++e) {
    int p  = e >> 1;
    int kl = ((p & 4) ? 16 : 0) + ((p & 3) << 1) + (half << 3) + (e & 1);
    a[e] = (__bf16)row[kk + kl];
  }
  return a;
}

// B-matrix 32x16 bf16: lane holds column N = lane%16.
// element e: K = (e>=8 ? 16 : 0) + (lane>=16 ? 8 : 0) + (e&7).  Source: bf16 panel in LDS.
__device__ __forceinline__ v16bf load_b_frag_lds(const __bf16* col, int kk, int half) {
  v16bf b;
#pragma unroll
  for (int e = 0; e < 16; ++e) b[e] = col[kk + ((e & 8) ? 16 : 0) + half * 8 + (e & 7)];
  return b;
}

__device__ __forceinline__ float sigmoidf_fast(float x) {
  return 1.0f / (1.0f + __expf(-x));
}
__device__ __forceinline__ float softplusf(float x) {
  return (x > 20.0f) ? x : log1pf(expf(x));
}

#define WMMA_BF16(A, Bf, Cacc) \
  __builtin_amdgcn_wmma_f32_16x16x32_bf16(false, (A), false, (Bf), (short)0, (Cacc), false, false)

// ---------------- Kernel 1: in_proj GEMM ----------------
// xz[m,e] = sum_c x[m,c] * W[e,c]   M=9216, N=512, K=128
// block = 128 m-rows x 64 n-cols; wave = 16x64 (4 accumulators)
#define INP_K 128
#define LDST_IN (INP_K + 8)    // stride 136 bf16 = 68 DW; 68%64=4 -> no bank clash
__global__ void k_in_proj(const float* __restrict__ x,
                          const float* __restrict__ w,
                          float* __restrict__ xz) {
  __shared__ __bf16 lw[64 * LDST_IN];
  const int tid  = threadIdx.x;
  const int nblk = blockIdx.x & 7;          // 8 n-blocks of 64
  const int mblk = blockIdx.x >> 3;         // 72 m-blocks of 128
  for (int idx = tid; idx < 64 * INP_K; idx += 256) {
    int col = idx / INP_K, k = idx % INP_K;
    lw[col * LDST_IN + k] = (__bf16)w[(size_t)(nblk * 64 + col) * INP_K + k];
  }
  __syncthreads();

  const int lane = tid & 31, wave = tid >> 5;
  const int half = (lane >> 4) & 1, n = lane & 15;
  const float* arow = x + (size_t)(mblk * 128 + wave * 16 + n) * INP_K;
  v8f acc0 = {}, acc1 = {}, acc2 = {}, acc3 = {};
#pragma unroll
  for (int kk = 0; kk < INP_K; kk += 32) {
    v16bf a = load_a_frag(arow, kk, half);
    acc0 = WMMA_BF16(a, load_b_frag_lds(lw + (n)      * LDST_IN, kk, half), acc0);
    acc1 = WMMA_BF16(a, load_b_frag_lds(lw + (16 + n) * LDST_IN, kk, half), acc1);
    acc2 = WMMA_BF16(a, load_b_frag_lds(lw + (32 + n) * LDST_IN, kk, half), acc2);
    acc3 = WMMA_BF16(a, load_b_frag_lds(lw + (48 + n) * LDST_IN, kk, half), acc3);
  }
#pragma unroll
  for (int v = 0; v < 8; ++v) {
    size_t m = mblk * 128 + wave * 16 + v + half * 8;
    float* o = xz + m * (2 * DD) + nblk * 64 + n;
    o[0] = acc0[v]; o[16] = acc1[v]; o[32] = acc2[v]; o[48] = acc3[v];
  }
}

// ---------------- Kernel 2: depthwise 3x3 conv + bias + SiLU ----------------
__global__ void k_conv(const float* __restrict__ xz,
                       const float* __restrict__ cw,
                       const float* __restrict__ cb,
                       float* __restrict__ xc) {
  int t = blockIdx.x * blockDim.x + threadIdx.x;
  int dch = t & (DD - 1);
  int rest = t >> 8;
  int w = rest % WW;  rest /= WW;
  int h = rest % HH;
  int b = rest / HH;

  float acc = 0.0f;
#pragma unroll
  for (int ky = 0; ky < 3; ++ky) {
    int hh2 = h + ky - 1;
    if (hh2 < 0 || hh2 >= HH) continue;
#pragma unroll
    for (int kx = 0; kx < 3; ++kx) {
      int ww2 = w + kx - 1;
      if (ww2 < 0 || ww2 >= WW) continue;
      float v = xz[((size_t)((b * HH + hh2) * WW + ww2)) * (2 * DD) + dch];
      acc += v * cw[dch * 9 + ky * 3 + kx];
    }
  }
  float s = acc + cb[dch];
  xc[t] = s * sigmoidf_fast(s);
}

// ---------------- Kernel 3: x_proj GEMM (4 directions) ----------------
// M = B*K*L = 36864; N = 40 (48 padded); K = 256
// block = 128 m-rows x 48 n-cols; wave = 16x48 (3 accumulators)
#define XP_K 256
#define LDST_XP (XP_K + 8)     // 264 bf16 = 132 DW; 132%64=4
__global__ void k_x_proj(const float* __restrict__ xc,
                         const float* __restrict__ wproj,
                         float* __restrict__ x_dbl) {
  __shared__ __bf16 lw[48 * LDST_XP];
  const int tid  = threadIdx.x;
  const int mblk = blockIdx.x;
  const int r0   = mblk * 128;              // 128 | 2304: (b,k) uniform in block
  const int b    = r0 / (KK * LL);
  const int k    = (r0 / LL) % KK;

  const float* wk = wproj + (size_t)k * CPROJ * XP_K;
  for (int idx = tid; idx < 48 * XP_K; idx += 256) {
    int col = idx / XP_K, kx = idx % XP_K;
    int csafe = (col < CPROJ) ? col : (CPROJ - 1);      // clamp: no OOB
    lw[col * LDST_XP + kx] = (__bf16)wk[(size_t)csafe * XP_K + kx];
  }
  __syncthreads();

  const int lane = tid & 31, wave = tid >> 5;
  const int half = (lane >> 4) & 1, n = lane & 15;
  int l = (r0 + wave * 16 + n) % LL;
  int pos;
  if (k == 0)      pos = l;
  else if (k == 1) pos = (l % HH) * WW + (l / HH);
  else if (k == 2) pos = LL - 1 - l;
  else { int l2 = LL - 1 - l; pos = (l2 % HH) * WW + (l2 / HH); }
  const float* arow = xc + ((size_t)b * LL + pos) * DD;

  v8f acc0 = {}, acc1 = {}, acc2 = {};
#pragma unroll
  for (int kk = 0; kk < XP_K; kk += 32) {
    v16bf a = load_a_frag(arow, kk, half);
    acc0 = WMMA_BF16(a, load_b_frag_lds(lw + (n)      * LDST_XP, kk, half), acc0);
    acc1 = WMMA_BF16(a, load_b_frag_lds(lw + (16 + n) * LDST_XP, kk, half), acc1);
    acc2 = WMMA_BF16(a, load_b_frag_lds(lw + (32 + n) * LDST_XP, kk, half), acc2);
  }
#pragma unroll
  for (int v = 0; v < 8; ++v) {
    size_t rs = r0 + wave * 16 + v + half * 8;
    float* o = x_dbl + rs * CPROJ;
    o[n]      = acc0[v];
    o[16 + n] = acc1[v];
    if (32 + n < CPROJ) o[32 + n] = acc2[v];
  }
}

// ---------------- Kernel 4a: chunked selective scan, local pass ----------------
// grid = (b,k,chunk) = 256 blocks; 256 threads = d channels
// Within chunk: h starts at 0; record sum(dt) and local h_end.
// Key identity: prod_l exp(dt_l*A) = exp(A * sum_l dt_l)
__global__ void k_scan_chunk(const float* __restrict__ xc,
                             const float* __restrict__ x_dbl,
                             const float* __restrict__ dtw,
                             const float* __restrict__ dtb,
                             const float* __restrict__ A_logs,
                             const float* __restrict__ Ds,
                             float* __restrict__ ys,
                             float* __restrict__ hend,
                             float* __restrict__ sumdt) {
  const int bk = blockIdx.x / SCHUNK;
  const int c  = blockIdx.x % SCHUNK;
  const int b  = bk / KK;
  const int k  = bk % KK;
  const int dch = threadIdx.x;
  const int kd  = k * DD + dch;

  float hst[NN];
  float Arow[NN];
#pragma unroll
  for (int n2 = 0; n2 < NN; ++n2) {
    hst[n2] = 0.0f;
    Arow[n2] = -expf(A_logs[(size_t)kd * NN + n2]);
  }
  float Dv = Ds[kd];
  float w8[RR];
#pragma unroll
  for (int r2 = 0; r2 < RR; ++r2) w8[r2] = dtw[(size_t)kd * RR + r2];
  float bias = dtb[kd];

  const size_t ybase = (size_t)bk * LL;
  const int l0 = c * CLEN;
  float sdt = 0.0f;

  for (int li = 0; li < CLEN; ++li) {
    int l = l0 + li;
    int pos;
    if (k == 0)      pos = l;
    else if (k == 1) pos = (l % HH) * WW + (l / HH);
    else if (k == 2) pos = LL - 1 - l;
    else { int l2 = LL - 1 - l; pos = (l2 % HH) * WW + (l2 / HH); }

    float u = xc[((size_t)b * LL + pos) * DD + dch];
    const float* row = x_dbl + (ybase + l) * CPROJ;
    __builtin_prefetch(row + CPROJ, 0, 1);           // gfx1250 global_prefetch

    float draw = bias;
#pragma unroll
    for (int r2 = 0; r2 < RR; ++r2) draw += w8[r2] * row[r2];
    float dt  = softplusf(draw);
    sdt += dt;
    float dtu = dt * u;

    float y = 0.0f;
#pragma unroll
    for (int n2 = 0; n2 < NN; ++n2) {
      float dA = __expf(dt * Arow[n2]);
      hst[n2] = hst[n2] * dA + dtu * row[RR + n2];
      y += hst[n2] * row[RR + NN + n2];
    }
    ys[(ybase + l) * DD + dch] = y + Dv * u;
  }

  const size_t sb = ((size_t)bk * SCHUNK + c) * DD + dch;
  sumdt[sb] = sdt;
#pragma unroll
  for (int n2 = 0; n2 < NN; ++n2) hend[sb * NN + n2] = hst[n2];
}

// ---------------- Kernel 4b: chunk-prefix recurrence ----------------
// one thread per (b,k,d,n): h_in[c+1] = exp(A*sumdt[c]) * h_in[c] + hend[c]
__global__ void k_scan_fix(const float* __restrict__ hend,
                           const float* __restrict__ sumdt,
                           const float* __restrict__ A_logs,
                           float* __restrict__ hin) {
  int t = blockIdx.x * blockDim.x + threadIdx.x;     // 65536 total
  int n2  = t & (NN - 1);
  int dch = (t >> 4) & (DD - 1);
  int bk  = t >> 12;
  int k   = bk % KK;
  float An = -expf(A_logs[((size_t)(k * DD + dch)) * NN + n2]);

  float h = 0.0f;
  for (int c = 0; c < SCHUNK; ++c) {
    size_t sb = ((size_t)bk * SCHUNK + c) * DD + dch;
    hin[sb * NN + n2] = h;
    h = __expf(An * sumdt[sb]) * h + hend[sb * NN + n2];
  }
}

// ---------------- Kernel 4c: apply cross-chunk state to outputs ----------------
// y_l += C_l . ( exp(A * s_l) * h_in ),  s_l = running sum of dt within chunk
__global__ void k_scan_apply(const float* __restrict__ x_dbl,
                             const float* __restrict__ dtw,
                             const float* __restrict__ dtb,
                             const float* __restrict__ A_logs,
                             const float* __restrict__ hin,
                             float* __restrict__ ys) {
  const int bk = blockIdx.x / SCHUNK;
  const int c  = blockIdx.x % SCHUNK;
  if (c == 0) return;                                // h_in == 0 for chunk 0
  const int k  = bk % KK;
  const int dch = threadIdx.x;
  const int kd  = k * DD + dch;

  float Arow[NN], hv[NN];
  const size_t sb = ((size_t)bk * SCHUNK + c) * DD + dch;
#pragma unroll
  for (int n2 = 0; n2 < NN; ++n2) {
    Arow[n2] = -expf(A_logs[(size_t)kd * NN + n2]);
    hv[n2]   = hin[sb * NN + n2];
  }
  float w8[RR];
#pragma unroll
  for (int r2 = 0; r2 < RR; ++r2) w8[r2] = dtw[(size_t)kd * RR + r2];
  float bias = dtb[kd];

  const size_t ybase = (size_t)bk * LL;
  const int l0 = c * CLEN;
  float s = 0.0f;

  for (int li = 0; li < CLEN; ++li) {
    int l = l0 + li;
    const float* row = x_dbl + (ybase + l) * CPROJ;
    __builtin_prefetch(row + CPROJ, 0, 1);

    float draw = bias;
#pragma unroll
    for (int r2 = 0; r2 < RR; ++r2) draw += w8[r2] * row[r2];
    s += softplusf(draw);

    float yadd = 0.0f;
#pragma unroll
    for (int n2 = 0; n2 < NN; ++n2)
      yadd += row[RR + NN + n2] * __expf(Arow[n2] * s) * hv[n2];
    ys[(ybase + l) * DD + dch] += yadd;
  }
}

// ---------------- Kernel 5: combine 4 dirs + LayerNorm + SiLU(z) gate --------
__global__ void k_combine(const float* __restrict__ ys,
                          const float* __restrict__ xz,
                          const float* __restrict__ g,
                          const float* __restrict__ beta,
                          float* __restrict__ yact) {
  __shared__ float s1[DD];
  __shared__ float s2[DD];
  const int m = blockIdx.x;
  const int b = m / LL;
  const int l0 = m % LL;
  const int dch = threadIdx.x;
  const int hh2 = l0 / WW;
  const int ww2 = l0 % WW;
  const int l1 = ww2 * HH + hh2;

  const size_t bb4 = (size_t)b * KK;
  float v = ys[((bb4 + 0) * LL + l0) * DD + dch]
          + ys[((bb4 + 2) * LL + (LL - 1 - l0)) * DD + dch]
          + ys[((bb4 + 1) * LL + l1) * DD + dch]
          + ys[((bb4 + 3) * LL + (LL - 1 - l1)) * DD + dch];

  s1[dch] = v;
  s2[dch] = v * v;
  __syncthreads();
  for (int st = DD / 2; st > 0; st >>= 1) {
    if (dch < st) { s1[dch] += s1[dch + st]; s2[dch] += s2[dch + st]; }
    __syncthreads();
  }
  float mean = s1[0] * (1.0f / DD);
  float var  = s2[0] * (1.0f / DD) - mean * mean;
  float yn = (v - mean) * rsqrtf(var + 1e-5f) * g[dch] + beta[dch];

  float z = xz[(size_t)m * (2 * DD) + DD + dch];
  yact[(size_t)m * DD + dch] = yn * z * sigmoidf_fast(z);
}

// ---------------- Kernel 6: out_proj GEMM ----------------
// out[m,c] = sum_d yact[m,d] * W[c,d]   M=9216, N=128, K=256
#define LDST_OP (XP_K + 8)
__global__ void k_out_proj(const float* __restrict__ yact,
                           const float* __restrict__ w,
                           float* __restrict__ out) {
  __shared__ __bf16 lw[64 * LDST_OP];
  const int tid  = threadIdx.x;
  const int nblk = blockIdx.x & 1;          // 2 n-blocks of 64
  const int mblk = blockIdx.x >> 1;         // 72 m-blocks of 128
  for (int idx = tid; idx < 64 * XP_K; idx += 256) {
    int col = idx / XP_K, k = idx % XP_K;
    lw[col * LDST_OP + k] = (__bf16)w[(size_t)(nblk * 64 + col) * XP_K + k];
  }
  __syncthreads();

  const int lane = tid & 31, wave = tid >> 5;
  const int half = (lane >> 4) & 1, n = lane & 15;
  const float* arow = yact + (size_t)(mblk * 128 + wave * 16 + n) * DD;
  v8f acc0 = {}, acc1 = {}, acc2 = {}, acc3 = {};
#pragma unroll
  for (int kk = 0; kk < XP_K; kk += 32) {
    v16bf a = load_a_frag(arow, kk, half);
    acc0 = WMMA_BF16(a, load_b_frag_lds(lw + (n)      * LDST_OP, kk, half), acc0);
    acc1 = WMMA_BF16(a, load_b_frag_lds(lw + (16 + n) * LDST_OP, kk, half), acc1);
    acc2 = WMMA_BF16(a, load_b_frag_lds(lw + (32 + n) * LDST_OP, kk, half), acc2);
    acc3 = WMMA_BF16(a, load_b_frag_lds(lw + (48 + n) * LDST_OP, kk, half), acc3);
  }
#pragma unroll
  for (int v = 0; v < 8; ++v) {
    size_t m = mblk * 128 + wave * 16 + v + half * 8;
    float* o = out + m * CC + nblk * 64 + n;
    o[0] = acc0[v]; o[16] = acc1[v]; o[32] = acc2[v]; o[48] = acc3[v];
  }
}

// ---------------- host launcher ----------------
extern "C" void kernel_launch(void* const* d_in, const int* in_sizes, int n_in,
                              void* d_out, int out_size, void* d_ws, size_t ws_size,
                              hipStream_t stream) {
  const float* x        = (const float*)d_in[0];
  const float* in_w     = (const float*)d_in[1];
  const float* conv_w   = (const float*)d_in[2];
  const float* conv_b   = (const float*)d_in[3];
  const float* xproj_w  = (const float*)d_in[4];
  const float* dtproj_w = (const float*)d_in[5];
  const float* dtproj_b = (const float*)d_in[6];
  const float* A_logs   = (const float*)d_in[7];
  const float* Ds       = (const float*)d_in[8];
  const float* norm_g   = (const float*)d_in[9];
  const float* norm_b   = (const float*)d_in[10];
  const float* outp_w   = (const float*)d_in[11];
  float* out = (float*)d_out;

  float* ws = (float*)d_ws;
  const size_t M = (size_t)BB * LL;                 // 9216
  size_t off_xz   = 0;                              // M * 512
  size_t off_xc   = off_xz   + M * (2 * DD);
  size_t off_xdbl = off_xc   + M * DD;
  size_t off_ys   = off_xdbl + M * KK * CPROJ;
  size_t off_yact = off_ys   + M * KK * DD;
  size_t off_hend = off_yact + M * DD;              // B*K*S*D*N = 1,048,576
  size_t off_hin  = off_hend + (size_t)BB * KK * SCHUNK * DD * NN;
  size_t off_sdt  = off_hin  + (size_t)BB * KK * SCHUNK * DD * NN;

  float* xz    = ws + off_xz;
  float* xc    = ws + off_xc;
  float* x_dbl = ws + off_xdbl;
  float* ysbuf = ws + off_ys;
  float* yact  = ws + off_yact;
  float* hend  = ws + off_hend;
  float* hin   = ws + off_hin;
  float* sdt   = ws + off_sdt;

  // 1) in_proj: 72 m-blocks x 8 n-blocks
  k_in_proj<<<dim3(72 * 8), dim3(256), 0, stream>>>(x, in_w, xz);
  // 2) depthwise conv + SiLU
  k_conv<<<dim3((BB * HH * WW * DD) / 256), dim3(256), 0, stream>>>(xz, conv_w, conv_b, xc);
  // 3) x_proj: 288 m-blocks (full 48-col panel per block)
  k_x_proj<<<dim3(288), dim3(256), 0, stream>>>(xc, xproj_w, x_dbl);
  // 4) chunk-parallel scan: local pass, prefix fix-up, apply
  k_scan_chunk<<<dim3(BB * KK * SCHUNK), dim3(DD), 0, stream>>>(
      xc, x_dbl, dtproj_w, dtproj_b, A_logs, Ds, ysbuf, hend, sdt);
  k_scan_fix<<<dim3((BB * KK * DD * NN) / 256), dim3(256), 0, stream>>>(
      hend, sdt, A_logs, hin);
  k_scan_apply<<<dim3(BB * KK * SCHUNK), dim3(DD), 0, stream>>>(
      x_dbl, dtproj_w, dtproj_b, A_logs, hin, ysbuf);
  // 5) combine + LN + gate
  k_combine<<<dim3(BB * LL), dim3(DD), 0, stream>>>(ysbuf, xz, norm_g, norm_b, yact);
  // 6) out_proj: 72 m-blocks x 2 n-blocks
  k_out_proj<<<dim3(72 * 2), dim3(256), 0, stream>>>(yact, outp_w, out);
}
